// GPT_47665547051045
// MI455X (gfx1250) — matrix operand order repaired and checked by reading, not compiled, not run on hardware
//
#include <hip/hip_runtime.h>

// ---------------------------------------------------------------------------
// GPT forward for gfx1250 (MI455X). All GEMMs + attention matmuls use
// v_wmma_f32_16x16x32_bf16 (bf16 in, fp32 accumulate). Weights are transposed
// + cast to bf16 [N][K] once per launch, so GEMM staging is pure contiguous
// 16B copies done with GLOBAL_LOAD_ASYNC_TO_LDS_B128 (ASYNCcnt-tracked,
// double-buffered, branch-free steady-state loop).
// ---------------------------------------------------------------------------

#define GPT_L 6
#define GPT_E 1024
#define GPT_H 16
#define GPT_D 64
#define GPT_T 1024
#define GPT_B 4
#define GPT_M (GPT_B * GPT_T)   // 4096 token rows
#define GPT_FF 4096
#define GPT_V 32000
#define GPT_QKVN (3 * GPT_E)    // 3072

typedef __attribute__((ext_vector_type(16))) __bf16 v16bf;
typedef __attribute__((ext_vector_type(8)))  float  v8f;

union FragU { int4 q[2]; v16bf v; };

static __device__ __forceinline__ unsigned short f2bf(float f) {
  unsigned int u = __float_as_uint(f);
  u += 0x7fffu + ((u >> 16) & 1u);        // round-to-nearest-even
  return (unsigned short)(u >> 16);
}

static __device__ __forceinline__ v8f wmma_bf16(const FragU& a, const FragU& b, v8f c) {
  return __builtin_amdgcn_wmma_f32_16x16x32_bf16(false, a.v, false, b.v,
                                                 (short)0, c, false, false);
}

// --------------------- transpose-cast weight preparation -------------------
// src[R][C] fp32 -> dst[C][R] bf16, 32x32 tiles through LDS (coalesced both ways)

static __device__ __forceinline__ void transpose_tile_f32_bf16(
    const float* __restrict__ src, unsigned short* __restrict__ dst,
    int R, int C, int r0, int c0, unsigned short (*tile)[33]) {
  const int tx = threadIdx.x & 31, ty = threadIdx.x >> 5;   // 32 x 8
#pragma unroll
  for (int i = 0; i < 32; i += 8)
    tile[ty + i][tx] = f2bf(src[(size_t)(r0 + ty + i) * C + c0 + tx]);
  __syncthreads();
#pragma unroll
  for (int i = 0; i < 32; i += 8)
    dst[(size_t)(c0 + ty + i) * R + r0 + tx] = tile[tx][ty + i];
}

__global__ __launch_bounds__(256) void transpose_cast_kernel(
    const float* __restrict__ src, unsigned short* __restrict__ dst,
    int R, int C, size_t srcStride, size_t dstStride) {
  __shared__ unsigned short tile[32][33];
  transpose_tile_f32_bf16(src + srcStride * blockIdx.z, dst + dstStride * blockIdx.z,
                          R, C, blockIdx.y * 32, blockIdx.x * 32, tile);
}

// Wq/Wk/Wv [L,H,E,D] -> bf16 [L][3072][E] (row n = r*1024+h*64+d, cols = e)
// i.e. 288 transposes of E x D blocks.
__global__ __launch_bounds__(256) void qkv_wcast_t_kernel(
    const float* __restrict__ Wq, const float* __restrict__ Wk,
    const float* __restrict__ Wv, unsigned short* __restrict__ dst) {
  __shared__ unsigned short tile[32][33];
  const int bz = blockIdx.z;                 // l*48 + r*16 + h
  const int l = bz / 48, rem = bz % 48, r = rem / 16, h = rem % 16;
  const float* W = (r == 0) ? Wq : (r == 1) ? Wk : Wv;
  const float* s = W + ((size_t)(l * GPT_H + h) * GPT_E) * GPT_D;
  unsigned short* d = dst + ((size_t)l * GPT_QKVN + r * GPT_E + h * GPT_D) * GPT_E;
  transpose_tile_f32_bf16(s, d, GPT_E, GPT_D, blockIdx.y * 32, blockIdx.x * 32, tile);
}

// bq/bk/bv [L,H,D] -> fp32 [L][3072]
__global__ __launch_bounds__(256) void qkv_bcat_kernel(
    const float* __restrict__ bq, const float* __restrict__ bk,
    const float* __restrict__ bv, float* __restrict__ dst) {
  int i = blockIdx.x * 256 + threadIdx.x;
  if (i >= GPT_L * GPT_QKVN) return;
  int n = i % GPT_QKVN, l = i / GPT_QKVN;
  int r = n >> 10, hd = n & 1023;
  dst[i] = ((r == 0) ? bq : (r == 1) ? bk : bv)[l * 1024 + hd];
}

// ------------------------------- embedding ---------------------------------

__global__ __launch_bounds__(256) void embed_kernel(
    const int* __restrict__ x, const float* __restrict__ wte,
    const float* __restrict__ wpe, unsigned short* __restrict__ h) {
  int row = blockIdx.x;                 // b*T + t
  int t = row & (GPT_T - 1);
  int tok = x[row];
  const float* wt = wte + (size_t)tok * GPT_E;
  const float* wp = wpe + (size_t)t * GPT_E;
  for (int e = threadIdx.x; e < GPT_E; e += 256)
    h[(size_t)row * GPT_E + e] = f2bf(wt[e] + wp[e]);
}

// --------------------------- WMMA bf16 GEMM --------------------------------
// C[M,N] = A[M,K] * Bt[N,K]^T (+bias, opt relu). Block tile 128x128, 8 waves,
// wave tile 32x64 (2x4 wmma 16x16x32). Both tiles staged to LDS with
// GLOBAL_LOAD_ASYNC_TO_LDS_B128, double-buffered on ASYNCcnt. Last K-tile is
// peeled so the steady-state loop is branch-free.

#define BM 128
#define BN 128
#define BK 32
#define AST 40
#define BST 40

__global__ __launch_bounds__(256) void gemm_bf16_kernel(
    const unsigned short* __restrict__ A, const unsigned short* __restrict__ Bt,
    const float* __restrict__ bias, float* __restrict__ Cf,
    unsigned short* __restrict__ Cb, int N, int K, int doRelu) {
  __shared__ unsigned short As[2][BM * AST];
  __shared__ unsigned short Bs[2][BN * BST];
  const int tid = threadIdx.x;
  const int lane = tid & 31, wave = tid >> 5;
  const int wr = wave >> 1, wc = wave & 1;
  const int hi = lane >> 4, l15 = lane & 15;
  const int m0 = blockIdx.y * BM, n0 = blockIdx.x * BN;

  // staging coordinates: chunks of 8 bf16 (16B); rows r and r+64
  const int sr = tid >> 2;              // 0..63
  const int sc = (tid & 3) << 3;        // 0,8,16,24

  const unsigned short* aRow0 = A + (size_t)(m0 + sr) * K + sc;
  const unsigned short* bRow0 = Bt + (size_t)(n0 + sr) * K + sc;

  auto stage = [&](int buf, int k0) {
    unsigned ldsA = (unsigned)(size_t)&As[buf][0] + (unsigned)(sr * AST + sc) * 2;
    unsigned ldsB = (unsigned)(size_t)&Bs[buf][0] + (unsigned)(sr * BST + sc) * 2;
    const unsigned short* ga = aRow0 + k0;
    const unsigned short* gb = bRow0 + k0;
    asm volatile("global_load_async_to_lds_b128 %0, %1, off"
                 :: "v"(ldsA), "v"(ga) : "memory");
    asm volatile("global_load_async_to_lds_b128 %0, %1, off"
                 :: "v"((unsigned)(ldsA + 64u * AST * 2u)), "v"(ga + (size_t)64 * K)
                 : "memory");
    asm volatile("global_load_async_to_lds_b128 %0, %1, off"
                 :: "v"(ldsB), "v"(gb) : "memory");
    asm volatile("global_load_async_to_lds_b128 %0, %1, off"
                 :: "v"((unsigned)(ldsB + 64u * BST * 2u)), "v"(gb + (size_t)64 * K)
                 : "memory");
    asm volatile("s_wait_xcnt 0x0" ::: "memory");   // addr VGPRs reusable
  };

  v8f acc[2][4];
#pragma unroll
  for (int i = 0; i < 2; ++i)
#pragma unroll
    for (int j = 0; j < 4; ++j)
#pragma unroll
      for (int e = 0; e < 8; ++e) acc[i][j][e] = 0.0f;

  auto compute = [&](int buf) {
    const unsigned short* as = As[buf];
    const unsigned short* bs = Bs[buf];
    FragU af[2], bfr[4];
#pragma unroll
    for (int ti = 0; ti < 2; ++ti) {    // A-layout: lane<16 K{0..7,16..23}, lane>=16 +8
      int row = wr * 32 + ti * 16 + l15;
      int kb = hi ? 8 : 0;
      af[ti].q[0] = *(const int4*)(as + row * AST + kb);
      af[ti].q[1] = *(const int4*)(as + row * AST + kb + 16);
    }
#pragma unroll
    for (int tj = 0; tj < 4; ++tj) {    // B-layout: lane<16 K{0..7,16..23}, lane>=16 +8
      int col = wc * 64 + tj * 16 + l15;
      int kb = hi ? 8 : 0;
      bfr[tj].q[0] = *(const int4*)(bs + col * BST + kb);
      bfr[tj].q[1] = *(const int4*)(bs + col * BST + kb + 16);
    }
#pragma unroll
    for (int ti = 0; ti < 2; ++ti)
#pragma unroll
      for (int tj = 0; tj < 4; ++tj)
        acc[ti][tj] = wmma_bf16(af[ti], bfr[tj], acc[ti][tj]);
  };

  const int nt = K / BK;                // always >= 2 here (K >= 1024)
  stage(0, 0);
  for (int it = 0; it < nt - 1; ++it) { // branch-free steady state
    const int buf = it & 1;
    stage(buf ^ 1, (it + 1) * BK);
    asm volatile("s_wait_asynccnt 0x4" ::: "memory");  // this tile done, next in flight
    __syncthreads();
    compute(buf);
    __syncthreads();                    // buf free before it is restaged
  }
  asm volatile("s_wait_asynccnt 0x0" ::: "memory");    // peeled last tile
  __syncthreads();
  compute((nt - 1) & 1);

#pragma unroll
  for (int ti = 0; ti < 2; ++ti)
#pragma unroll
    for (int tj = 0; tj < 4; ++tj) {
      int col = n0 + wc * 64 + tj * 16 + l15;
      float bv = bias ? bias[col] : 0.0f;
#pragma unroll
      for (int e = 0; e < 8; ++e) {     // C-layout: lane<16 rows v, lane>=16 rows v+8
        int row = m0 + wr * 32 + ti * 16 + e + hi * 8;
        float val = acc[ti][tj][e] + bv;
        if (doRelu) val = fmaxf(val, 0.0f);
        size_t idx = (size_t)row * N + col;
        if (Cf) Cf[idx] = val;
        if (Cb) Cb[idx] = f2bf(val);
      }
    }
}

// qkv[M,3072] -> q[b,h,t,d], k[b,h,t,d], vT[b,h,d,t] (all bf16)
__global__ __launch_bounds__(256) void qkv_repack_kernel(
    const unsigned short* __restrict__ qkv, unsigned short* __restrict__ q,
    unsigned short* __restrict__ k, unsigned short* __restrict__ vT) {
  size_t i = (size_t)blockIdx.x * 256 + threadIdx.x;
  if (i >= (size_t)GPT_M * GPT_QKVN) return;
  int n = (int)(i % GPT_QKVN);
  int mrow = (int)(i / GPT_QKVN);
  int b = mrow >> 10, t = mrow & 1023;
  int r = n >> 10, hd = n & 1023, h = hd >> 6, d = hd & 63;
  int bh = b * GPT_H + h;
  unsigned short val = qkv[i];
  if (r == 0)      q[((size_t)bh * GPT_T + t) * GPT_D + d] = val;
  else if (r == 1) k[((size_t)bh * GPT_T + t) * GPT_D + d] = val;
  else             vT[((size_t)bh * GPT_D + d) * GPT_T + t] = val;
}

// --------------------- flash attention (WMMA, causal) ----------------------
// One wave per (b,h,16-query tile). S = Q*K^T via 2 chained wmmas (D=64),
// online softmax (16-lane shuffle reductions per C-layout half), P bounced
// through LDS (C-layout -> A-layout), O += P*V via 4 wmmas per 32-key block.

__global__ __launch_bounds__(128) void attn_fa_kernel(
    const unsigned short* __restrict__ Q, const unsigned short* __restrict__ Km,
    const unsigned short* __restrict__ Vt, unsigned short* __restrict__ O) {
  __shared__ unsigned short Ps[4][16 * 40];
  const int lane = threadIdx.x & 31, wave = threadIdx.x >> 5;
  const int hi = lane >> 4, l15 = lane & 15;
  const int gw = blockIdx.x * 4 + wave;
  const int tilesPer = GPT_T / 16;        // 64
  const int bh = gw / tilesPer;
  const int qt = gw % tilesPer;
  const int b = bh / GPT_H, h = bh % GPT_H;
  const int qBase = qt * 16;
  const unsigned short* qp = Q + (size_t)bh * GPT_T * GPT_D;
  const unsigned short* kp = Km + (size_t)bh * GPT_T * GPT_D;
  const unsigned short* vp = Vt + (size_t)bh * GPT_D * GPT_T;
  unsigned short* ps = Ps[wave];

  FragU qa[2];
  {
    const unsigned short* qrow = qp + (size_t)(qBase + l15) * GPT_D;
    int kb = hi ? 8 : 0;
    qa[0].q[0] = *(const int4*)(qrow + kb);
    qa[0].q[1] = *(const int4*)(qrow + kb + 16);
    qa[1].q[0] = *(const int4*)(qrow + 32 + kb);
    qa[1].q[1] = *(const int4*)(qrow + 32 + kb + 16);
  }

  v8f o[4];
  float m[8], ls[8];
#pragma unroll
  for (int dt = 0; dt < 4; ++dt)
#pragma unroll
    for (int e = 0; e < 8; ++e) o[dt][e] = 0.0f;
#pragma unroll
  for (int e = 0; e < 8; ++e) { m[e] = -1e30f; ls[e] = 0.0f; }

  const int nblk = (qBase + 16 + 31) / 32;
  for (int blk = 0; blk < nblk; ++blk) {
    const int kb0 = blk * 32;
    v8f s[2];
#pragma unroll
    for (int sj = 0; sj < 2; ++sj)
#pragma unroll
      for (int e = 0; e < 8; ++e) s[sj][e] = 0.0f;

#pragma unroll
    for (int sj = 0; sj < 2; ++sj) {    // B = K^T: lane n = key, contiguous d
      const unsigned short* krow = kp + (size_t)(kb0 + sj * 16 + l15) * GPT_D;
      int db = hi ? 8 : 0;
      FragU kf0, kf1;
      kf0.q[0] = *(const int4*)(krow + db);
      kf0.q[1] = *(const int4*)(krow + db + 16);
      kf1.q[0] = *(const int4*)(krow + 32 + db);
      kf1.q[1] = *(const int4*)(krow + 32 + db + 16);
      s[sj] = wmma_bf16(qa[0], kf0, s[sj]);
      s[sj] = wmma_bf16(qa[1], kf1, s[sj]);
    }

#pragma unroll
    for (int e = 0; e < 8; ++e) {       // scale, causal mask, online softmax
      const int row = qBase + e + hi * 8;
      float a0 = s[0][e] * 0.125f;      // D^-0.5
      float a1 = s[1][e] * 0.125f;
      if (kb0 + l15 > row)       a0 = -1e30f;
      if (kb0 + 16 + l15 > row)  a1 = -1e30f;
      float mx = fmaxf(a0, a1);
      mx = fmaxf(mx, __shfl_xor(mx, 1));
      mx = fmaxf(mx, __shfl_xor(mx, 2));
      mx = fmaxf(mx, __shfl_xor(mx, 4));
      mx = fmaxf(mx, __shfl_xor(mx, 8));
      float mn = fmaxf(m[e], mx);
      float scOld = __expf(m[e] - mn);
      float p0 = __expf(a0 - mn);
      float p1 = __expf(a1 - mn);
      float rs = p0 + p1;
      rs += __shfl_xor(rs, 1);
      rs += __shfl_xor(rs, 2);
      rs += __shfl_xor(rs, 4);
      rs += __shfl_xor(rs, 8);
      ls[e] = ls[e] * scOld + rs;
      m[e] = mn;
      s[0][e] = p0;
      s[1][e] = p1;
#pragma unroll
      for (int dt = 0; dt < 4; ++dt) o[dt][e] *= scOld;
    }

    // P: C-layout fp32 -> LDS row-major bf16 -> A-layout fragment
#pragma unroll
    for (int sj = 0; sj < 2; ++sj)
#pragma unroll
      for (int e = 0; e < 8; ++e)
        ps[(e + hi * 8) * 40 + sj * 16 + l15] = f2bf(s[sj][e]);
    asm volatile("s_wait_dscnt 0" ::: "memory");

    FragU pf;
    {
      int kb = hi ? 8 : 0;
      pf.q[0] = *(const int4*)(ps + l15 * 40 + kb);
      pf.q[1] = *(const int4*)(ps + l15 * 40 + kb + 16);
    }

#pragma unroll
    for (int dt = 0; dt < 4; ++dt) {    // B = V: vT gives lane n = d, contiguous keys
      const unsigned short* vrow = vp + (size_t)(dt * 16 + l15) * GPT_T + kb0;
      int koff = hi ? 8 : 0;
      FragU vf;
      vf.q[0] = *(const int4*)(vrow + koff);
      vf.q[1] = *(const int4*)(vrow + koff + 16);
      o[dt] = wmma_bf16(pf, vf, o[dt]);
    }
  }

#pragma unroll
  for (int e = 0; e < 8; ++e) {         // O -> [b, t, h*64+d] bf16 (heads concat)
    float inv = 1.0f / ls[e];
    int row = qBase + e + hi * 8;
    size_t base = ((size_t)b * GPT_T + row) * GPT_E + h * GPT_D + l15;
#pragma unroll
    for (int dt = 0; dt < 4; ++dt)
      O[base + dt * 16] = f2bf(o[dt][e] * inv);
  }
}

// ------------------------------- layernorm ---------------------------------

__global__ __launch_bounds__(256) void layernorm_bf16_kernel(
    const float* __restrict__ X, const float* __restrict__ g,
    const float* __restrict__ bta, unsigned short* __restrict__ Y) {
  const int lane = threadIdx.x & 31, wave = threadIdx.x >> 5;
  const int row = blockIdx.x * 8 + wave;
  const float* x = X + (size_t)row * GPT_E;
  float s = 0.0f, s2 = 0.0f;
  for (int i = lane; i < GPT_E; i += 32) { float v = x[i]; s += v; s2 += v * v; }
#pragma unroll
  for (int msk = 1; msk < 32; msk <<= 1) {
    s += __shfl_xor(s, msk);
    s2 += __shfl_xor(s2, msk);
  }
  const float mean = s * (1.0f / GPT_E);
  const float var = s2 * (1.0f / GPT_E) - mean * mean;
  const float inv = rsqrtf(var + 1e-5f);
  unsigned short* y = Y + (size_t)row * GPT_E;
  for (int i = lane; i < GPT_E; i += 32)
    y[i] = f2bf((x[i] - mean) * inv * g[i] + bta[i]);
}

// --------------------------------- host ------------------------------------

extern "C" void kernel_launch(void* const* d_in, const int* in_sizes, int n_in,
                              void* d_out, int out_size, void* d_ws, size_t ws_size,
                              hipStream_t stream) {
  (void)in_sizes; (void)n_in; (void)out_size; (void)ws_size;

  const int*   x   = (const int*)d_in[0];
  const float* wte = (const float*)d_in[1];
  const float* wpe = (const float*)d_in[2];
  const float* Wq  = (const float*)d_in[3];
  const float* bq  = (const float*)d_in[4];
  const float* Wk  = (const float*)d_in[5];
  const float* bk  = (const float*)d_in[6];
  const float* Wv  = (const float*)d_in[7];
  const float* bv  = (const float*)d_in[8];
  const float* Wp  = (const float*)d_in[9];
  const float* bp  = (const float*)d_in[10];
  const float* g1  = (const float*)d_in[11];
  const float* be1 = (const float*)d_in[12];
  const float* W1  = (const float*)d_in[13];
  const float* bf1 = (const float*)d_in[14];
  const float* W2  = (const float*)d_in[15];
  const float* bf2 = (const float*)d_in[16];
  const float* g2  = (const float*)d_in[17];
  const float* be2 = (const float*)d_in[18];
  const float* lmW = (const float*)d_in[19];
  const float* lmb = (const float*)d_in[20];
  float* out = (float*)d_out;

  char* ws = (char*)d_ws;
  size_t off = 0;
  auto take = [&](size_t bytes) -> char* {
    char* p = ws + off;
    off = (off + bytes + 255) & ~(size_t)255;
    return p;
  };
  // ~343 MB total workspace; all weight copies are in transposed [N][K] bf16.
  unsigned short* wqkv_t = (unsigned short*)take((size_t)GPT_L * GPT_QKVN * GPT_E * 2);
  float*          bqkv   = (float*)         take((size_t)GPT_L * GPT_QKVN * 4);
  unsigned short* wp_t   = (unsigned short*)take((size_t)GPT_L * GPT_E * GPT_E * 2);
  unsigned short* w1_t   = (unsigned short*)take((size_t)GPT_L * GPT_FF * GPT_E * 2);
  unsigned short* w2_t   = (unsigned short*)take((size_t)GPT_L * GPT_E * GPT_FF * 2);
  unsigned short* lmw_t  = (unsigned short*)take((size_t)GPT_V * GPT_E * 2);
  unsigned short* h_b    = (unsigned short*)take((size_t)GPT_M * GPT_E * 2);
  unsigned short* qkv_b  = (unsigned short*)take((size_t)GPT_M * GPT_QKVN * 2);
  unsigned short* q_b    = (unsigned short*)take((size_t)GPT_B * GPT_H * GPT_T * GPT_D * 2);
  unsigned short* k_b    = (unsigned short*)take((size_t)GPT_B * GPT_H * GPT_T * GPT_D * 2);
  unsigned short* vT_b   = (unsigned short*)take((size_t)GPT_B * GPT_H * GPT_D * GPT_T * 2);
  unsigned short* o_b    = (unsigned short*)take((size_t)GPT_M * GPT_E * 2);
  unsigned short* h1_b   = (unsigned short*)take((size_t)GPT_M * GPT_E * 2);
  unsigned short* f_b    = (unsigned short*)take((size_t)GPT_M * GPT_FF * 2);
  float*          tmp_f  = (float*)         take((size_t)GPT_M * GPT_E * 4);

  // ---- weight preparation: transpose + cast to bf16 [N][K] ----
  qkv_wcast_t_kernel<<<dim3(GPT_D / 32, GPT_E / 32, GPT_L * 3 * GPT_H), 256, 0, stream>>>(
      Wq, Wk, Wv, wqkv_t);
  qkv_bcat_kernel<<<(GPT_L * GPT_QKVN + 255) / 256, 256, 0, stream>>>(bq, bk, bv, bqkv);
  transpose_cast_kernel<<<dim3(GPT_E / 32, GPT_E / 32, GPT_L), 256, 0, stream>>>(
      Wp, wp_t, GPT_E, GPT_E, (size_t)GPT_E * GPT_E, (size_t)GPT_E * GPT_E);
  transpose_cast_kernel<<<dim3(GPT_FF / 32, GPT_E / 32, GPT_L), 256, 0, stream>>>(
      W1, w1_t, GPT_E, GPT_FF, (size_t)GPT_E * GPT_FF, (size_t)GPT_E * GPT_FF);
  transpose_cast_kernel<<<dim3(GPT_E / 32, GPT_FF / 32, GPT_L), 256, 0, stream>>>(
      W2, w2_t, GPT_FF, GPT_E, (size_t)GPT_FF * GPT_E, (size_t)GPT_FF * GPT_E);
  transpose_cast_kernel<<<dim3(GPT_V / 32, GPT_E / 32, 1), 256, 0, stream>>>(
      lmW, lmw_t, GPT_E, GPT_V, 0, 0);

  // ---- embedding ----
  embed_kernel<<<GPT_M, 256, 0, stream>>>(x, wte, wpe, h_b);

  const dim3 gemm_blk(256);
  for (int l = 0; l < GPT_L; ++l) {
    // QKV projection: [4096,1024] x [1024,3072]
    gemm_bf16_kernel<<<dim3(GPT_QKVN / BN, GPT_M / BM), gemm_blk, 0, stream>>>(
        h_b, wqkv_t + (size_t)l * GPT_QKVN * GPT_E, bqkv + (size_t)l * GPT_QKVN,
        nullptr, qkv_b, GPT_QKVN, GPT_E, 0);
    qkv_repack_kernel<<<(unsigned)(((size_t)GPT_M * GPT_QKVN + 255) / 256), 256, 0, stream>>>(
        qkv_b, q_b, k_b, vT_b);

    // causal flash attention: 4096 wave-tiles, 4 waves/block
    attn_fa_kernel<<<(GPT_B * GPT_H * (GPT_T / 16)) / 4, 128, 0, stream>>>(q_b, k_b, vT_b, o_b);

    // output projection -> fp32, then LN1 -> bf16
    gemm_bf16_kernel<<<dim3(GPT_E / BN, GPT_M / BM), gemm_blk, 0, stream>>>(
        o_b, wp_t + (size_t)l * GPT_E * GPT_E, bp + (size_t)l * GPT_E,
        tmp_f, nullptr, GPT_E, GPT_E, 0);
    layernorm_bf16_kernel<<<GPT_M / 8, 256, 0, stream>>>(
        tmp_f, g1 + (size_t)l * GPT_E, be1 + (size_t)l * GPT_E, h1_b);

    // FF1 (relu) -> bf16
    gemm_bf16_kernel<<<dim3(GPT_FF / BN, GPT_M / BM), gemm_blk, 0, stream>>>(
        h1_b, w1_t + (size_t)l * GPT_FF * GPT_E, bf1 + (size_t)l * GPT_FF,
        nullptr, f_b, GPT_FF, GPT_E, 1);

    // FF2 (relu) -> fp32, then LN2 -> bf16 (next layer input)
    gemm_bf16_kernel<<<dim3(GPT_E / BN, GPT_M / BM), gemm_blk, 0, stream>>>(
        f_b, w2_t + (size_t)l * GPT_E * GPT_FF, bf2 + (size_t)l * GPT_E,
        tmp_f, nullptr, GPT_E, GPT_FF, 1);
    layernorm_bf16_kernel<<<GPT_M / 8, 256, 0, stream>>>(
        tmp_f, g2 + (size_t)l * GPT_E, be2 + (size_t)l * GPT_E, h_b);
  }

  // LM head: [4096,1024] x [1024,32000] -> fp32 logits
  gemm_bf16_kernel<<<dim3(GPT_V / BN, GPT_M / BM), gemm_blk, 0, stream>>>(
      h_b, lmw_t, lmb, out, nullptr, GPT_V, GPT_E, 0);
}
